// ATTLayer_76390288327094
// MI455X (gfx1250) — matrix operand order repaired and checked by reading, not compiled, not run on hardware
//
#include <hip/hip_runtime.h>
#include <hip/hip_bf16.h>

typedef __attribute__((ext_vector_type(16))) __bf16 v16bf;
typedef __attribute__((ext_vector_type(8)))  float  v8f;

#define BATCH  64
#define SEQ    512
#define NH     1024
#define NA     256
#define NROWS  (BATCH * SEQ)   // 32768
#define KSTEPS (NH / 32)       // 32
#define NTILES (NA / 16)       // 16
#define LDS_B_BYTES (NTILES * 32 * 32)   // 16 KB of packed B per K-step

// Async global->LDS copy of 16 bytes per lane (VFLAT/VGLOBAL encoding, GV
// mode: VDST = per-lane LDS byte address, VADDR = 64-bit global address,
// SADDR = off). Tracked by ASYNCcnt.
__device__ __forceinline__ void async_copy_b128(void* lds_dst, const void* gsrc) {
  unsigned           lds_off = (unsigned)(unsigned long long)(uintptr_t)lds_dst;
  unsigned long long gaddr   = (unsigned long long)(uintptr_t)gsrc;
  asm volatile("global_load_async_to_lds_b128 %0, %1, off"
               :
               : "v"(lds_off), "v"(gaddr)
               : "memory");
}

__device__ __forceinline__ void wait_async_zero() {
#if __has_builtin(__builtin_amdgcn_s_wait_asynccnt)
  __builtin_amdgcn_s_wait_asynccnt(0);
#else
  asm volatile("s_wait_asynccnt 0x0" ::: "memory");
#endif
}

// -------------------------------------------------------------------------
// Kernel 1: fold bias into W and pre-pack into the WMMA B-matrix lane layout.
// Entry index = ((kstep*NTILES + ntile)*32 + lane), 16 bf16 (32B) per entry.
// Lane l (0..15): column N = ntile*16 + l, element e holds K = kstep*32 + e.
// Lane l (16..31): same column, element e holds K = kstep*32 + 16 + e.
// -------------------------------------------------------------------------
__global__ void __launch_bounds__(256)
prep_wb_kernel(const float* __restrict__ W, const float* __restrict__ bias,
               __bf16* __restrict__ wbp) {
  int idx = blockIdx.x * blockDim.x + threadIdx.x;
  if (idx >= KSTEPS * NTILES * 32) return;
  int lane  = idx & 31;
  int nt    = (idx >> 5) & (NTILES - 1);
  int kstep = idx >> 9;
  int col   = lane & 15;
  int n     = nt * 16 + col;
  int kbase = kstep * 32 + ((lane < 16) ? 0 : 16);
  float bv  = bias[n];
  __bf16* dst = wbp + (size_t)idx * 16;
#pragma unroll
  for (int e = 0; e < 16; ++e) {
    dst[e] = (__bf16)(W[(size_t)(kbase + e) * NA + n] + bv);
  }
}

// -------------------------------------------------------------------------
// Kernel 2: fused GEMM (bf16 WMMA, f32 accum) + tanh + dot with u -> logits.
// One wave per 16-row tile of X; 16 N-tiles accumulated across K=1024.
// B slab (16 KB per K-step) is staged once per block into double-buffered
// LDS via async global->LDS copies (ASYNCcnt), feeding WMMA from ds_load.
// -------------------------------------------------------------------------
__global__ void __launch_bounds__(256)
gemm_tanh_logit_kernel(const float* __restrict__ x,
                       const v16bf* __restrict__ wbp,
                       const float* __restrict__ u,
                       float* __restrict__ logit) {
  __shared__ __align__(16) unsigned char lds_b[2][LDS_B_BYTES];

  int tid     = threadIdx.x;
  int lane    = tid & 31;
  int wave    = tid >> 5;
  int rowTile = blockIdx.x * 8 + wave;          // 0 .. 2047
  int row     = rowTile * 16 + (lane & 15);     // A-matrix: M = lane & 15
  const float* xrow = x + (size_t)row * NH;
  int kb = (lane < 16) ? 0 : 16;                // A lane layout K split

  const char* gB = (const char*)wbp;            // packed B base

  // Stage B[kstep] (16 KB, contiguous) into lds_b[buf]: 4 x 16B per thread.
  auto stageB = [&](int kstep, int buf) {
    const char* gsrc = gB + (size_t)kstep * LDS_B_BYTES + tid * 16;
    char*       ldst = (char*)&lds_b[buf][0] + tid * 16;
#pragma unroll
    for (int i = 0; i < 4; ++i) {
      async_copy_b128(ldst + i * 4096, gsrc + i * 4096);
    }
  };

  v8f acc[NTILES];
#pragma unroll
  for (int nt = 0; nt < NTILES; ++nt) {
    v8f z = {0.f, 0.f, 0.f, 0.f, 0.f, 0.f, 0.f, 0.f};
    acc[nt] = z;
  }

  stageB(0, 0);
  wait_async_zero();
  __syncthreads();

  for (int kstep = 0; kstep < KSTEPS; ++kstep) {
    int cur = kstep & 1;
    if (kstep + 1 < KSTEPS) stageB(kstep + 1, cur ^ 1);  // prefetch next slab

    // A K order used here (matches packed B): lanes 0-15 hold K {0..15},
    // lanes 16-31 hold K {16..31} of this 32-wide K chunk.
    const float* p = xrow + kstep * 32 + kb;
    float4 f0 = *(const float4*)(p);
    float4 f1 = *(const float4*)(p + 4);
    float4 f2 = *(const float4*)(p + 8);
    float4 f3 = *(const float4*)(p + 12);
    v16bf a;
    a[0]  = (__bf16)f0.x; a[1]  = (__bf16)f0.y; a[2]  = (__bf16)f0.z; a[3]  = (__bf16)f0.w;
    a[4]  = (__bf16)f1.x; a[5]  = (__bf16)f1.y; a[6]  = (__bf16)f1.z; a[7]  = (__bf16)f1.w;
    a[8]  = (__bf16)f2.x; a[9]  = (__bf16)f2.y; a[10] = (__bf16)f2.z; a[11] = (__bf16)f2.w;
    a[12] = (__bf16)f3.x; a[13] = (__bf16)f3.y; a[14] = (__bf16)f3.z; a[15] = (__bf16)f3.w;

    const v16bf* bl = (const v16bf*)&lds_b[cur][0];
#pragma unroll
    for (int nt = 0; nt < NTILES; ++nt) {
      v16bf bm = bl[nt * 32 + lane];               // ds_load_b128 x2
      acc[nt] = __builtin_amdgcn_wmma_f32_16x16x32_bf16(
          /*neg_a=*/false, a, /*neg_b=*/false, bm,
          /*c_mod=*/(short)0, acc[nt], /*reuse_a=*/false, /*reuse_b=*/false);
    }

    // Drain this wave's async copies, then block-barrier: next iteration's
    // reads (and buffer overwrites) are safe only after both.
    wait_async_zero();
    __syncthreads();
  }

  // Epilogue: logit[m] = sum_n tanh(acc[m][n]) * u[n]
  // C/D layout: lane holds column N = (lane&15), rows M = r + 8*(lane>=16).
  int col = lane & 15;
  float part[8];
#pragma unroll
  for (int r = 0; r < 8; ++r) part[r] = 0.f;
#pragma unroll
  for (int nt = 0; nt < NTILES; ++nt) {
    float uval = u[nt * 16 + col];
#pragma unroll
    for (int r = 0; r < 8; ++r) part[r] += tanhf(acc[nt][r]) * uval;
  }
  // Reduce over the 16 lanes of each half-wave (wave32: masks 1,2,4,8).
#pragma unroll
  for (int r = 0; r < 8; ++r) {
    float v = part[r];
    v += __shfl_xor(v, 1, 32);
    v += __shfl_xor(v, 2, 32);
    v += __shfl_xor(v, 4, 32);
    v += __shfl_xor(v, 8, 32);
    part[r] = v;
  }
  if (col == 0) {                       // lanes 0 and 16 write their 8 rows
    int mbase = (lane >> 4) * 8;
#pragma unroll
    for (int r = 0; r < 8; ++r)
      logit[rowTile * 16 + mbase + r] = part[r];
  }
}

// -------------------------------------------------------------------------
// Kernel 3: softmax over seq (one block of 512 threads per batch row)
// -------------------------------------------------------------------------
__global__ void __launch_bounds__(512)
softmax_kernel(const float* __restrict__ logit, float* __restrict__ weights) {
  int b   = blockIdx.x;
  int tid = threadIdx.x;
  __shared__ float red[16];
  __shared__ float bmax;
  __shared__ float bsum;

  float v = logit[b * SEQ + tid];
  float m = v;
#pragma unroll
  for (int o = 16; o >= 1; o >>= 1) m = fmaxf(m, __shfl_xor(m, o, 32));
  if ((tid & 31) == 0) red[tid >> 5] = m;
  __syncthreads();
  if (tid == 0) {
    float t = red[0];
    for (int i = 1; i < 16; ++i) t = fmaxf(t, red[i]);
    bmax = t;
  }
  __syncthreads();

  float e = expf(v - bmax);
  float s = e;
#pragma unroll
  for (int o = 16; o >= 1; o >>= 1) s += __shfl_xor(s, o, 32);
  if ((tid & 31) == 0) red[tid >> 5] = s;
  __syncthreads();
  if (tid == 0) {
    float t = 0.f;
    for (int i = 0; i < 16; ++i) t += red[i];
    bsum = t;
  }
  __syncthreads();

  weights[b * SEQ + tid] = e / bsum;
}

// -------------------------------------------------------------------------
// Kernel 4: att_out[b,h] = sum_s weights[b,s] * x[b,s,h]
// -------------------------------------------------------------------------
__global__ void __launch_bounds__(256)
wsum_kernel(const float* __restrict__ x, const float* __restrict__ w,
            float* __restrict__ out) {
  int b = blockIdx.x;
  int h = blockIdx.y * 256 + threadIdx.x;
  __shared__ float ws[SEQ];
  for (int s = threadIdx.x; s < SEQ; s += 256) ws[s] = w[b * SEQ + s];
  __syncthreads();
  const float* xb = x + (size_t)b * SEQ * NH + h;
  float acc = 0.f;
#pragma unroll 4
  for (int s = 0; s < SEQ; ++s) acc += ws[s] * xb[(size_t)s * NH];
  out[b * NH + h] = acc;
}

// -------------------------------------------------------------------------
extern "C" void kernel_launch(void* const* d_in, const int* in_sizes, int n_in,
                              void* d_out, int out_size, void* d_ws, size_t ws_size,
                              hipStream_t stream) {
  const float* x    = (const float*)d_in[0];  // [64,512,1024]
  const float* W    = (const float*)d_in[1];  // [1024,256]
  const float* u    = (const float*)d_in[2];  // [256,1]
  const float* bias = (const float*)d_in[3];  // [256]
  float* out = (float*)d_out;                 // [64,1024]

  char* ws = (char*)d_ws;
  __bf16* wbp    = (__bf16*)ws;                                // 512 KB packed W+b
  float*  logits = (float*)(ws + 512 * 1024);                  // 128 KB
  float*  wts    = (float*)(ws + 512 * 1024 + 128 * 1024);     // 128 KB

  prep_wb_kernel<<<(KSTEPS * NTILES * 32 + 255) / 256, 256, 0, stream>>>(W, bias, wbp);
  gemm_tanh_logit_kernel<<<NROWS / 16 / 8, 256, 0, stream>>>(
      x, (const v16bf*)wbp, u, logits);
  softmax_kernel<<<BATCH, 512, 0, stream>>>(logits, wts);
  wsum_kernel<<<dim3(BATCH, NH / 256), 256, 0, stream>>>(x, wts, out);
}
